// FP8LinearSoftmax_43061342110431
// MI455X (gfx1250) — 8-bit matrix operand order repaired and checked by reading, compiled, not run on hardware
//
#include <hip/hip_runtime.h>
#include <hip/hip_fp16.h>
#include <stdint.h>


#define NUM_BH   96
#define SEQ      1024
#define HDIM     64
#define NHEADS   12
#define KTOP     204
#define ROWS     (NUM_BH * SEQ)
#define ROWS_PER_HEAD 8192.0f
#define ROWSTRIDE 1040

#if defined(__has_builtin)
# if __has_builtin(__builtin_amdgcn_cvt_pk_fp8_f32)
#  define HAVE_PK_FP8 1
# endif
# if __has_builtin(__builtin_amdgcn_cvt_f32_fp8)
#  define HAVE_F32_FP8 1
# endif
#endif
#ifndef HAVE_PK_FP8
# define HAVE_PK_FP8 0
#endif
#ifndef HAVE_F32_FP8
# define HAVE_F32_FP8 0
#endif

typedef __attribute__((ext_vector_type(16))) int   v16i;
typedef __attribute__((ext_vector_type(8)))  float v8f;


__device__ __forceinline__ float h16(float x) {
  return __half2float(__float2half(x));
}

__device__ __forceinline__ float clamp448(float x) {
  return fminf(fmaxf(x, -448.0f), 448.0f);
}

__device__ __forceinline__ uint32_t sw_fp8_encode(float x) {
  uint32_t sgn = (__float_as_uint(x) >> 31) << 7;
  float ax = fminf(fabsf(x), 448.0f);
  if (!(ax > 0.0f)) return 0u;
  int e; frexpf(ax, &e); e -= 1;
  uint32_t code;
  if (e < -6) {
    int md = (int)rintf(ldexpf(ax, 9));
    code = (uint32_t)md;
  } else {
    int mi = (int)rintf(ldexpf(ax, (float)(3 - e)));
    if (mi >= 16) { mi = 8; e += 1; }
    if (e > 8)    { e = 8; mi = 14; }
    code = (uint32_t)(((e + 7) << 3) | (mi - 8));
  }
  if (code > 0x7Eu) code = 0x7Eu;
  return sgn | code;
}

__device__ __forceinline__ float sw_fp8_decode(uint32_t b) {
  uint32_t mag = b & 0x7Fu, ef = mag >> 3, mf = mag & 7u;
  float v = (ef == 0u) ? ldexpf((float)mf, -9)
                       : ldexpf((float)(8u + mf), (int)ef - 10);
  return (b & 0x80u) ? -v : v;
}

__device__ __forceinline__ uint32_t pk_encode4(float a, float b, float c, float d) {
#if HAVE_PK_FP8
  int p = __builtin_amdgcn_cvt_pk_fp8_f32(a, b, 0, false);
  p = __builtin_amdgcn_cvt_pk_fp8_f32(c, d, p, true);
  return (uint32_t)p;
#else
  return sw_fp8_encode(a) | (sw_fp8_encode(b) << 8) |
         (sw_fp8_encode(c) << 16) | (sw_fp8_encode(d) << 24);
#endif
}

template <int SEL>
__device__ __forceinline__ float byte_decode(uint32_t w) {
#if HAVE_F32_FP8
  return __builtin_amdgcn_cvt_f32_fp8((int)w, SEL);
#else
  return sw_fp8_decode((w >> (8 * SEL)) & 0xFFu);
#endif
}

__device__ __forceinline__ float decode_sum4(uint32_t w) {
  return byte_decode<0>(w) + byte_decode<1>(w) +
         byte_decode<2>(w) + byte_decode<3>(w);
}

__device__ __forceinline__ float fp8q(float x) {
#if HAVE_PK_FP8 && HAVE_F32_FP8
  int p = __builtin_amdgcn_cvt_pk_fp8_f32(clamp448(x), 0.0f, 0, false);
  return __builtin_amdgcn_cvt_f32_fp8(p, 0);
#else
  return sw_fp8_decode(sw_fp8_encode(x));
#endif
}

__device__ __forceinline__ uint32_t fp8_key(uint32_t b) {
  uint32_t mag = b & 0x7Fu;
  return (b & 0x80u) ? (127u - mag) : (128u + mag);
}


__global__ void __launch_bounds__(256)
quant_topk_kernel(const float* __restrict__ aw, uint8_t* __restrict__ aw8,
                  int* __restrict__ rowTh, int* __restrict__ rowNeed,
                  float* __restrict__ headsum) {
  __shared__ int hist[8][256];
  __shared__ int tot[256];
  const int row = blockIdx.x, tid = threadIdx.x;
  const int wave = tid >> 5;
#pragma unroll
  for (int w = 0; w < 8; ++w) hist[w][tid] = 0;
  __syncthreads();

  float4 x = ((const float4*)(aw + (size_t)row * SEQ))[tid];
  const uint32_t packed = pk_encode4(clamp448(x.x), clamp448(x.y),
                                     clamp448(x.z), clamp448(x.w));
#pragma unroll
  for (int j = 0; j < 4; ++j)
    atomicAdd(&hist[wave][fp8_key((packed >> (8 * j)) & 0xFFu)], 1);
  ((uint32_t*)aw8)[row * 256 + tid] = packed;
  __syncthreads();

  int t = 0;
#pragma unroll
  for (int w = 0; w < 8; ++w) t += hist[w][tid];
  tot[tid] = t;
  __syncthreads();

  if (tid == 0) {
    int acc = 0, th = 0;
    for (int kk = 255; kk >= 0; --kk) {
      acc += tot[kk];
      if (acc >= KTOP) { th = kk; break; }
    }
    const int G = acc - tot[th];
    rowTh[row]   = th;
    rowNeed[row] = KTOP - G;
    uint32_t byte = (th >= 128) ? (uint32_t)(th - 128)
                                : (0x80u | (uint32_t)(127 - th));
    atomicAdd(&headsum[(row >> 10) % NHEADS], byte_decode<0>(byte));
  }
}


__global__ void __launch_bounds__(256)
packb_kernel(const float* __restrict__ v, uint32_t* __restrict__ packB) {
  const int idx  = blockIdx.x * 256 + threadIdx.x;
  const int d    = idx & 15;
  const int lane = (idx >> 4) & 31;
  const int ct   = (idx >> 9) & 3;
  const int ks   = (idx >> 11) & 7;
  const int bh   = idx >> 14;
  const int col  = ct * 16 + (lane & 15);
  const int hi   = (lane >> 4) * 8;
  const int g    = d >> 2, kq = d & 3;
  const int kbase = ks * 128 + (d >> 3) * 64 + ((d >> 1) & 3) * 16 + hi + (d & 1) * 4;
  const float* vb = v + ((size_t)bh * SEQ + kbase) * HDIM + col;
  packB[idx] = pk_encode4(clamp448(vb[0]), clamp448(vb[HDIM]),
                          clamp448(vb[2 * HDIM]), clamp448(vb[3 * HDIM]));
}


__device__ __forceinline__ v16i load_a_frag(const uint8_t* rowbase) {
  v16i a;
#pragma unroll
  for (int hh = 0; hh < 2; ++hh)
#pragma unroll
    for (int q = 0; q < 4; ++q) {
      uint2 w = *(const uint2*)(rowbase + hh * 64 + q * 16);
      a[hh * 8 + q * 2]     = (int)w.x;
      a[hh * 8 + q * 2 + 1] = (int)w.y;
    }
  return a;
}

__global__ void __launch_bounds__(256)
fused_gemm_kernel(const uint8_t* __restrict__ aw8, const int* __restrict__ rowTh,
                  const int* __restrict__ rowNeed,
                  const float* __restrict__ headsum,
                  const uint32_t* __restrict__ packB, float* __restrict__ out) {
  __shared__ uint8_t smE[32 * ROWSTRIDE];
  __shared__ uint8_t smS[32 * ROWSTRIDE];
  __shared__ uint8_t smM[32 * ROWSTRIDE];
  __shared__ float   denomSm[32];

  const int bh = blockIdx.x;
  const int tid = threadIdx.x;
  const int wave = tid >> 5, lane = tid & 31;
  const int rowBlock = bh * SEQ + blockIdx.y * 32;

  const int h = bh % NHEADS;
  const float mean = headsum[h] * (1.0f / ROWS_PER_HEAD);
  const float kh = expf(mean);
  const float bb = (1.0f - mean) * kh;

#pragma unroll
  for (int r = 0; r < 4; ++r) {
    const int local = wave * 4 + r;
    const int rowg  = rowBlock + local;
    const uint32_t th = (uint32_t)rowTh[rowg];
    const int need = rowNeed[rowg];

    const uint4* cp = (const uint4*)(aw8 + (size_t)rowg * SEQ);
    uint4 c0 = cp[lane * 2], c1 = cp[lane * 2 + 1];
    uint32_t cw[8] = {c0.x, c0.y, c0.z, c0.w, c1.x, c1.y, c1.z, c1.w};

    int eqc = 0;
#pragma unroll
    for (int j = 0; j < 8; ++j)
#pragma unroll
      for (int b2 = 0; b2 < 4; ++b2)
        eqc += (fp8_key((cw[j] >> (8 * b2)) & 0xFFu) == th);

    int pre = eqc;
#pragma unroll
    for (int off = 1; off < 32; off <<= 1) {
      int n = __shfl_up(pre, off);
      if (lane >= off) pre += n;
    }
    int run = pre - eqc;

    uint32_t* pE = (uint32_t*)(smE + local * ROWSTRIDE) + lane * 8;
    uint32_t* pS = (uint32_t*)(smS + local * ROWSTRIDE) + lane * 8;
    uint32_t* pM = (uint32_t*)(smM + local * ROWSTRIDE) + lane * 8;
    float sumE = 0.0f, sumS = 0.0f;
#pragma unroll
    for (int j = 0; j < 8; ++j) {
      const uint32_t c4 = cw[j];
      uint32_t m4 = 0;
#pragma unroll
      for (int b2 = 0; b2 < 4; ++b2) {
        const uint32_t key = fp8_key((c4 >> (8 * b2)) & 0xFFu);
        const bool eq = (key == th);
        if ((key > th) || (eq && run < need)) m4 |= 0xFFu << (8 * b2);
        if (eq) run++;
      }
      const uint32_t sB = c4 & m4;
      const uint32_t mB = 0x38383838u & m4;
      float e0 = ((m4 >>  0) & 1u) ? 0.0f : clamp448(__expf(byte_decode<0>(c4)));
      float e1 = ((m4 >>  8) & 1u) ? 0.0f : clamp448(__expf(byte_decode<1>(c4)));
      float e2 = ((m4 >> 16) & 1u) ? 0.0f : clamp448(__expf(byte_decode<2>(c4)));
      float e3 = ((m4 >> 24) & 1u) ? 0.0f : clamp448(__expf(byte_decode<3>(c4)));
      const uint32_t eB = pk_encode4(e0, e1, e2, e3);
      sumS += decode_sum4(sB);
      sumE += decode_sum4(eB);
      pE[j] = eB; pS[j] = sB; pM[j] = mB;
    }
#pragma unroll
    for (int off = 16; off > 0; off >>= 1) {
      sumE += __shfl_down(sumE, off);
      sumS += __shfl_down(sumS, off);
    }
    if (lane == 0)
      denomSm[local] = h16(sumE + kh * sumS + bb * (float)KTOP);
  }
  __syncthreads();

  const int wrow = wave >> 2, ct = wave & 3;
  const int mrow = lane & 15;
  const int hiB  = (lane >> 4) * 8;

  const uint8_t* pe = smE + (wrow * 16 + mrow) * ROWSTRIDE + hiB;
  const uint8_t* ps = smS + (wrow * 16 + mrow) * ROWSTRIDE + hiB;
  const uint8_t* pm = smM + (wrow * 16 + mrow) * ROWSTRIDE + hiB;

  v8f accE = {}, accS = {}, accM = {};

#pragma unroll
  for (int ks = 0; ks < 8; ++ks) {
    const uint4* bp = (const uint4*)(packB + (((bh * 8 + ks) * 4 + ct) << 9)
                                     + lane * 16);
    uint4 b0 = bp[0], b1 = bp[1], b2 = bp[2], b3 = bp[3];
    v16i bf;
    bf[0]=(int)b0.x; bf[1]=(int)b0.y; bf[2]=(int)b0.z; bf[3]=(int)b0.w;
    bf[4]=(int)b1.x; bf[5]=(int)b1.y; bf[6]=(int)b1.z; bf[7]=(int)b1.w;
    bf[8]=(int)b2.x; bf[9]=(int)b2.y; bf[10]=(int)b2.z; bf[11]=(int)b2.w;
    bf[12]=(int)b3.x; bf[13]=(int)b3.y; bf[14]=(int)b3.z; bf[15]=(int)b3.w;

    v16i aE = load_a_frag(pe + ks * 128);
    v16i aS = load_a_frag(ps + ks * 128);
    v16i aM = load_a_frag(pm + ks * 128);

    accE = __builtin_amdgcn_wmma_f32_16x16x128_fp8_fp8(aE, bf, (short)0, accE,
                                                       false, false);
    accS = __builtin_amdgcn_wmma_f32_16x16x128_fp8_fp8(aS, bf, (short)0, accS,
                                                       false, false);
    accM = __builtin_amdgcn_wmma_f32_16x16x128_fp8_fp8(aM, bf, (short)0, accM,
                                                       false, false);
  }

  const int nlo = lane & 15, hi = lane >> 4;
#pragma unroll
  for (int i = 0; i < 8; ++i) {
    const int localRow = wrow * 16 + hi * 8 + i;
    const int rowg = rowBlock + localRow;
    const float dense = h16(accE[i]);
    const float s1 = h16(kh * fp8q(accS[i]));
    const float s2 = h16(bb * accM[i]);
    const float tot = h16(dense + h16(s1 + s2));
    out[(size_t)rowg * HDIM + ct * 16 + nlo] = fp8q(tot / denomSm[localRow]);
  }
}


extern "C" void kernel_launch(void* const* d_in, const int* in_sizes, int n_in,
                              void* d_out, int out_size, void* d_ws, size_t ws_size,
                              hipStream_t stream) {
  (void)in_sizes; (void)n_in; (void)out_size; (void)ws_size;
  const float* aw = (const float*)d_in[0];
  const float* v  = (const float*)d_in[1];
  float* out = (float*)d_out;

  const size_t SZ = (size_t)NUM_BH * SEQ * SEQ;
  uint8_t*  AW8    = (uint8_t*)d_ws;
  uint32_t* PACKB  = (uint32_t*)(AW8 + SZ);
  int*      ROWTH  = (int*)((uint8_t*)PACKB + (size_t)NUM_BH * 8 * 4 * 32 * 64);
  int*      ROWNEED= ROWTH + ROWS;
  float*    HEADSUM= (float*)(ROWNEED + ROWS);

  hipMemsetAsync(HEADSUM, 0, 16 * sizeof(float), stream);

  quant_topk_kernel<<<ROWS, 256, 0, stream>>>(aw, AW8, ROWTH, ROWNEED, HEADSUM);
  packb_kernel<<<(NUM_BH * 8 * 4 * 32 * 16) / 256, 256, 0, stream>>>(v, PACKB);
  fused_gemm_kernel<<<dim3(NUM_BH, 32), 256, 0, stream>>>(AW8, ROWTH, ROWNEED,
                                                          HEADSUM, PACKB, out);
}
